// cnf_62508954026195
// MI455X (gfx1250) — matrix operand order repaired and checked by reading, not compiled, not run on hardware
//
#include <hip/hip_runtime.h>
#include <hip/hip_bf16.h>
#include <math.h>

// ---------------------------------------------------------------------------
// Problem constants (from reference): N=8192, IN=128, H0=512, H1=256, ITER=10
// ---------------------------------------------------------------------------
static constexpr int N_    = 8192;
static constexpr int IN_   = 128;
static constexpr int H0_   = 512;
static constexpr int H1_   = 256;
static constexpr int ITER_ = 10;

typedef __bf16 bf16;
typedef __attribute__((ext_vector_type(8)))  bf16  bf16x8;
typedef __attribute__((ext_vector_type(16))) bf16  bf16x16;
typedef __attribute__((ext_vector_type(8)))  float f32x8;
typedef __attribute__((ext_vector_type(4)))  int   i32x4;

// Async global->LDS staging (CDNA5 GLOBAL_LOAD_ASYNC_TO_LDS_B128, ASYNCcnt).
// Guarded: falls back to synchronous staging if the builtin is absent.
// Probe-confirmed (round 2 diagnostic): arg0 is a non-const addrspace(1)
// pointer to a 4 x i32 vector; arity is 4.
#if defined(__HIP_DEVICE_COMPILE__) && __has_builtin(__builtin_amdgcn_global_load_async_to_lds_b128)
#define USE_ASYNC_LDS 1
typedef __attribute__((address_space(1))) i32x4* g_i32x4_ptr;
typedef __attribute__((address_space(3))) i32x4* l_i32x4_ptr;
#else
#define USE_ASYNC_LDS 0
#endif

__device__ __forceinline__ unsigned short f32_to_bf16_rne(float f) {
  union { float f; unsigned int u; } v; v.f = f;
  unsigned int u = v.u;
  u += 0x7FFFu + ((u >> 16) & 1u);           // round-to-nearest-even
  return (unsigned short)(u >> 16);
}

// ---------------------------------------------------------------------------
// f32 -> bf16 conversion
// ---------------------------------------------------------------------------
__global__ void k_cvt_bf16(const float* __restrict__ in,
                           unsigned short* __restrict__ out, int n) {
  int i = blockIdx.x * blockDim.x + threadIdx.x;
  if (i < n) out[i] = f32_to_bf16_rne(in[i]);
}

// ---------------------------------------------------------------------------
// WMMA NT GEMM:  C[m,n] = sum_k A[m,k]*B[n,k] + bias[n]
// A: [M,K] row-major bf16, B: [N,K] row-major bf16, C: [M,ldc] f32.
// Block = 256 threads = 8 waves. Block tile 128(M) x 64(N); each wave owns a
// 16 x 64 strip = 4 accumulators. A/B fragments are loaded straight from
// global (inputs are tiny -> L2-resident) with a contiguous-K order shared by A and B:
//   A lane(l): rows m0+(l&15); 16 contiguous K starting at kh*16
//   B lane(l): col  n0+(l&15); 16 contiguous K starting at kh*16
//   D lane(l), vgpr p: (m0 + 8*kh + p, n0 + (l&15))
// ---------------------------------------------------------------------------
__global__ __launch_bounds__(256)
void k_wmma_gemm_nt_bias(const bf16* __restrict__ A, const bf16* __restrict__ B,
                         const float* __restrict__ bias, float* __restrict__ C,
                         int K, int ldc) {
  const int tid  = threadIdx.x;
  const int w    = tid >> 5;
  const int lane = tid & 31;
  const int lm   = lane & 15;
  const int kh   = lane >> 4;
  const int m0   = blockIdx.x * 128 + w * 16;
  const int n0   = blockIdx.y * 64;

  f32x8 zero = {0.f,0.f,0.f,0.f,0.f,0.f,0.f,0.f};
  f32x8 acc[4];
#pragma unroll
  for (int j = 0; j < 4; ++j) acc[j] = zero;

  const bf16* Arow = A + (size_t)(m0 + lm) * K;
  for (int kb = 0; kb < K; kb += 32) {
    bf16x8 alo = *(const bf16x8*)(Arow + kb + kh * 16);
    bf16x8 ahi = *(const bf16x8*)(Arow + kb + kh * 16 + 8);
    bf16x16 av = __builtin_shufflevector(alo, ahi,
        0,1,2,3,4,5,6,7,8,9,10,11,12,13,14,15);
    const bf16* Bp = B + (size_t)(n0 + lm) * K + kb + kh * 16;
#pragma unroll
    for (int j = 0; j < 4; ++j) {
      bf16x16 bv = *(const bf16x16*)(Bp + (size_t)j * 16 * K);
      acc[j] = __builtin_amdgcn_wmma_f32_16x16x32_bf16(
                   false, av, false, bv, (short)0, acc[j], false, false);
    }
  }

#pragma unroll
  for (int j = 0; j < 4; ++j) {
    int n = n0 + j * 16 + lm;
    float bn = bias[n];
#pragma unroll
    for (int p = 0; p < 8; ++p) {
      int m = m0 + 8 * kh + p;
      C[(size_t)m * ldc + n] = acc[j][p] + bn;
    }
  }
}

// ---------------------------------------------------------------------------
// BatchNorm column stats (biased var), folded into scale/shift:
//   y = z*scale[c] + shift[c]  with scale = g*rsqrt(var+eps), shift = be-mu*scale
// One block per column.
// ---------------------------------------------------------------------------
__global__ __launch_bounds__(256)
void k_col_stats(const float* __restrict__ Z, int rows, int ld,
                 const float* __restrict__ g, const float* __restrict__ be,
                 float* __restrict__ scale, float* __restrict__ shift) {
  __shared__ float s1[256], s2[256];
  const int col = blockIdx.x, tid = threadIdx.x;
  float a = 0.f, b = 0.f;
  for (int r = tid; r < rows; r += 256) {
    float v = Z[(size_t)r * ld + col];
    a += v; b += v * v;
  }
  s1[tid] = a; s2[tid] = b; __syncthreads();
  for (int o = 128; o; o >>= 1) {
    if (tid < o) { s1[tid] += s1[tid + o]; s2[tid] += s2[tid + o]; }
    __syncthreads();
  }
  if (tid == 0) {
    float mu  = s1[0] / rows;
    float var = s2[0] / rows - mu * mu;
    float rs  = rsqrtf(var + 1e-5f);
    float sc  = g[col] * rs;
    scale[col] = sc;
    shift[col] = be[col] - mu * sc;
  }
}

// Layer-1 BN + leaky_relu -> bf16 (elementwise; H0 is a power of two)
__global__ void k_bn_act_bf16(const float* __restrict__ Z,
                              const float* __restrict__ scale,
                              const float* __restrict__ shift,
                              unsigned short* __restrict__ out,
                              int total, int colmask) {
  int i = blockIdx.x * blockDim.x + threadIdx.x;
  if (i >= total) return;
  int c = i & colmask;
  float v = Z[i] * scale[c] + shift[c];
  v = v > 0.f ? v : 0.01f * v;
  out[i] = f32_to_bf16_rne(v);
}

// ---------------------------------------------------------------------------
// Layer-2 BN + leaky_relu, fused with per-row fc dot (logits0), row-norm
// rsqrt, and bf16 feat store. One block (256 thr) per row, H1 == 256.
// ---------------------------------------------------------------------------
__global__ __launch_bounds__(256)
void k_bn_act_row(const float* __restrict__ Z2, const float* __restrict__ scale,
                  const float* __restrict__ shift, const float* __restrict__ fcw,
                  const float* __restrict__ fcb,
                  unsigned short* __restrict__ featsb,
                  float* __restrict__ unary, float* __restrict__ lg,
                  float* __restrict__ invfn) {
  __shared__ float sd[256], sq[256];
  const int row = blockIdx.x, c = threadIdx.x;
  float v = Z2[(size_t)row * H1_ + c] * scale[c] + shift[c];
  v = v > 0.f ? v : 0.01f * v;
  featsb[(size_t)row * H1_ + c] = f32_to_bf16_rne(v);
  sd[c] = v * fcw[c]; sq[c] = v * v; __syncthreads();
  for (int o = 128; o; o >>= 1) {
    if (c < o) { sd[c] += sd[c + o]; sq[c] += sq[c + o]; }
    __syncthreads();
  }
  if (c == 0) {
    float u = sd[0] + fcb[0];
    unary[row] = u;
    lg[row]    = u;
    invfn[row] = rsqrtf(sq[0]);
  }
}

// ---------------------------------------------------------------------------
// PP = (feats@featsT)*invfn_i*invfn_j * 0.5*(W[i,j] + W[j,i])
// Big WMMA GEMM (K=256), fused epilogue. Block tile 128(M) x 64(N), 8 waves.
// W[i,j] is read directly (coalesced along N); the transposed block W[j,i]
// is staged into LDS with ASYNC global->LDS B128 transfers issued at kernel
// entry and waited (s_wait_asynccnt) only after the whole WMMA K-loop, so the
// 32KB/block transpose stage is hidden behind matrix math.
// ---------------------------------------------------------------------------
__global__ __launch_bounds__(256)
void k_sim_pp(const bf16* __restrict__ F, const float* __restrict__ invfn,
              const float* __restrict__ W, float* __restrict__ PP) {
  __shared__ float wt[64][129];                 // W[n0+r][m0b+c], padded
  const int tid = threadIdx.x;
  const int m0b = blockIdx.x * 128;
  const int n0  = blockIdx.y * 64;

#if USE_ASYNC_LDS
  // 64 rows x 128 f32 = 64 x (32 lanes x 16B). 2048 lane-transfers / 256 thr.
  for (int l = tid; l < 64 * 32; l += 256) {
    int r = l >> 5, c = (l & 31) * 4;
    __builtin_amdgcn_global_load_async_to_lds_b128(
        (g_i32x4_ptr)(W + (size_t)(n0 + r) * N_ + m0b + c),
        (l_i32x4_ptr)&wt[r][c],
        0, 0);
  }
#else
  for (int l = tid; l < 64 * 128; l += 256) {   // coalesced transpose stage
    int r = l >> 7, c = l & 127;
    wt[r][c] = W[(size_t)(n0 + r) * N_ + m0b + c];
  }
  __syncthreads();
#endif

  const int w = tid >> 5, lane = tid & 31, lm = lane & 15, kh = lane >> 4;
  const int m0 = m0b + w * 16;

  f32x8 zero = {0.f,0.f,0.f,0.f,0.f,0.f,0.f,0.f};
  f32x8 acc[4];
#pragma unroll
  for (int j = 0; j < 4; ++j) acc[j] = zero;

  const bf16* Arow = F + (size_t)(m0 + lm) * H1_;
#pragma unroll
  for (int kb = 0; kb < H1_; kb += 32) {
    bf16x8 alo = *(const bf16x8*)(Arow + kb + kh * 16);
    bf16x8 ahi = *(const bf16x8*)(Arow + kb + kh * 16 + 8);
    bf16x16 av = __builtin_shufflevector(alo, ahi,
        0,1,2,3,4,5,6,7,8,9,10,11,12,13,14,15);
    const bf16* Bp = F + (size_t)(n0 + lm) * H1_ + kb + kh * 16;
#pragma unroll
    for (int j = 0; j < 4; ++j) {
      bf16x16 bv = *(const bf16x16*)(Bp + (size_t)j * 16 * H1_);
      acc[j] = __builtin_amdgcn_wmma_f32_16x16x32_bf16(
                   false, av, false, bv, (short)0, acc[j], false, false);
    }
  }

#if USE_ASYNC_LDS
  // Drain this wave's async transfers, then barrier so every wave's slice of
  // wt[][] is visible before the epilogue reads it.
#if __has_builtin(__builtin_amdgcn_s_wait_asynccnt)
  __builtin_amdgcn_s_wait_asynccnt(0);
#else
  asm volatile("s_wait_asynccnt 0x0" ::: "memory");
#endif
  __syncthreads();
#endif

  float im[8];
#pragma unroll
  for (int p = 0; p < 8; ++p) im[p] = invfn[m0 + 8 * kh + p];

#pragma unroll
  for (int j = 0; j < 4; ++j) {
    int nl = j * 16 + lm;
    int n  = n0 + nl;
    float sn = invfn[n];
#pragma unroll
    for (int p = 0; p < 8; ++p) {
      int m = m0 + 8 * kh + p;
      float wsym = 0.5f * (W[(size_t)m * N_ + n] + wt[nl][w * 16 + 8 * kh + p]);
      PP[(size_t)m * N_ + n] = acc[j][p] * im[p] * sn * wsym;
    }
  }
}

// coef = 1 - 2*sigmoid(logits)
__global__ void k_coef(const float* __restrict__ lg, float* __restrict__ coef,
                       int n) {
  int i = blockIdx.x * blockDim.x + threadIdx.x;
  if (i < n) {
    float p = 1.0f / (1.0f + __expf(-lg[i]));
    coef[i] = 1.0f - 2.0f * p;
  }
}

// out = unary + PP @ coef   (one wave per row, float4 streaming, shfl reduce)
__global__ __launch_bounds__(256)
void k_matvec(const float* __restrict__ PP, const float* __restrict__ coef,
              const float* __restrict__ unary, float* __restrict__ out) {
  const int row  = blockIdx.x * 8 + (threadIdx.x >> 5);
  const int lane = threadIdx.x & 31;
  const float4* r  = (const float4*)(PP + (size_t)row * N_);
  const float4* c4 = (const float4*)coef;
  float s = 0.f;
  for (int t = lane; t < N_ / 4; t += 32) {
    float4 a = r[t], b = c4[t];
    s += a.x * b.x + a.y * b.y + a.z * b.z + a.w * b.w;
  }
  for (int o = 16; o; o >>= 1) s += __shfl_xor(s, o, 32);
  if (lane == 0) out[row] = unary[row] + s;
}

// ---------------------------------------------------------------------------
// Launch
// ---------------------------------------------------------------------------
extern "C" void kernel_launch(void* const* d_in, const int* in_sizes, int n_in,
                              void* d_out, int out_size, void* d_ws, size_t ws_size,
                              hipStream_t stream) {
  const float* x   = (const float*)d_in[0];
  const float* W   = (const float*)d_in[1];
  const float* W1  = (const float*)d_in[2];
  const float* b1  = (const float*)d_in[3];
  const float* g1  = (const float*)d_in[4];
  const float* be1 = (const float*)d_in[5];
  const float* W2  = (const float*)d_in[6];
  const float* b2  = (const float*)d_in[7];
  const float* g2  = (const float*)d_in[8];
  const float* be2 = (const float*)d_in[9];
  const float* fcw = (const float*)d_in[10];
  const float* fcb = (const float*)d_in[11];
  float* out = (float*)d_out;

  // Workspace carve-out (~295 MB; PP dominates at 256 MB)
  char* p = (char*)d_ws;
  auto carve = [&](size_t bytes) -> char* {
    char* q = p; p += (bytes + 255) & ~(size_t)255; return q;
  };
  float*          z1    = (float*)         carve((size_t)N_ * H0_ * 4);
  float*          z2    = (float*)         carve((size_t)N_ * H1_ * 4);
  unsigned short* xb    = (unsigned short*)carve((size_t)N_ * IN_ * 2);
  unsigned short* w1b   = (unsigned short*)carve((size_t)H0_ * IN_ * 2);
  unsigned short* w2b   = (unsigned short*)carve((size_t)H1_ * H0_ * 2);
  unsigned short* hb    = (unsigned short*)carve((size_t)N_ * H0_ * 2);
  unsigned short* fb    = (unsigned short*)carve((size_t)N_ * H1_ * 2);
  float*          sc1   = (float*)         carve(H0_ * 4);
  float*          sh1   = (float*)         carve(H0_ * 4);
  float*          sc2   = (float*)         carve(H1_ * 4);
  float*          sh2   = (float*)         carve(H1_ * 4);
  float*          invfn = (float*)         carve(N_ * 4);
  float*          unary = (float*)         carve(N_ * 4);
  float*          lgA   = (float*)         carve(N_ * 4);
  float*          lgB   = (float*)         carve(N_ * 4);
  float*          coef  = (float*)         carve(N_ * 4);
  float*          PP    = (float*)         carve((size_t)N_ * N_ * 4);
  (void)ws_size; (void)in_sizes; (void)n_in; (void)out_size;

  // bf16 operand staging
  k_cvt_bf16<<<(N_ * IN_) / 256, 256, 0, stream>>>(x,  xb,  N_ * IN_);
  k_cvt_bf16<<<(H0_ * IN_) / 256, 256, 0, stream>>>(W1, w1b, H0_ * IN_);
  k_cvt_bf16<<<(H1_ * H0_) / 256, 256, 0, stream>>>(W2, w2b, H1_ * H0_);

  // Layer 1: z1 = x@W1^T + b1 ; BN ; lrelu -> hb (bf16)
  k_wmma_gemm_nt_bias<<<dim3(N_ / 128, H0_ / 64), 256, 0, stream>>>(
      (const bf16*)xb, (const bf16*)w1b, b1, z1, IN_, H0_);
  k_col_stats<<<H0_, 256, 0, stream>>>(z1, N_, H0_, g1, be1, sc1, sh1);
  k_bn_act_bf16<<<(N_ * H0_) / 256, 256, 0, stream>>>(z1, sc1, sh1, hb,
                                                      N_ * H0_, H0_ - 1);

  // Layer 2: z2 = h@W2^T + b2 ; BN ; lrelu ; fused logits0 + row norms
  k_wmma_gemm_nt_bias<<<dim3(N_ / 128, H1_ / 64), 256, 0, stream>>>(
      (const bf16*)hb, (const bf16*)w2b, b2, z2, H0_, H1_);
  k_col_stats<<<H1_, 256, 0, stream>>>(z2, N_, H1_, g2, be2, sc2, sh2);
  k_bn_act_row<<<N_, 256, 0, stream>>>(z2, sc2, sh2, fcw, fcb, fb,
                                       unary, lgA, invfn);

  // PP = cosine-sim * W_sym  (big WMMA GEMM, fused epilogue, async W^T stage)
  k_sim_pp<<<dim3(N_ / 128, N_ / 64), 256, 0, stream>>>(
      (const bf16*)fb, invfn, W, PP);

  // 10 mean-field iterations: coef = 1-2*sigmoid(lg); lg = unary + PP@coef
  float* cur = lgA;
  float* nxt = lgB;
  for (int t = 0; t < ITER_; ++t) {
    k_coef<<<N_ / 256, 256, 0, stream>>>(cur, coef, N_);
    float* dst = (t == ITER_ - 1) ? out : nxt;
    k_matvec<<<N_ / 8, 256, 0, stream>>>(PP, coef, unary, dst);
    float* tmp = cur; cur = nxt; nxt = tmp;
  }
}
